// Net_60052232733176
// MI455X (gfx1250) — matrix operand order repaired and checked by reading, not compiled, not run on hardware
//
#include <hip/hip_runtime.h>
#include <hip/hip_bf16.h>
#include <cmath>
#include <cstring>

// Problem constants (match reference)
#define NNODES 50000
#define NEDGES 600000
#define NGRAPH 50
#define NPG    1000
#define DIM    128
#define NCLS   10
#define NEGV   (-1e30f)

typedef __attribute__((ext_vector_type(16))) __bf16 v16bf;
typedef __attribute__((ext_vector_type(8)))  float  v8f;

// ---------------- helpers ----------------
__device__ __forceinline__ unsigned d_f2ord(float f) {
  unsigned u = __float_as_uint(f);
  return (u & 0x80000000u) ? ~u : (u | 0x80000000u);
}
__device__ __forceinline__ float d_ord2f(unsigned u) {
  return (u & 0x80000000u) ? __uint_as_float(u & 0x7fffffffu) : __uint_as_float(~u);
}
static unsigned h_f2ord(float f) {
  unsigned u; std::memcpy(&u, &f, 4);
  return (u & 0x80000000u) ? ~u : (u | 0x80000000u);
}
__device__ __forceinline__ float d_lrelu02(float x) { return x > 0.f ? x : 0.2f * x; }

// ---------------- fill kernels ----------------
__global__ void k_fill_f32(float* p, float v, int n) {
  int i = blockIdx.x * blockDim.x + threadIdx.x;
  if (i < n) p[i] = v;
}
__global__ void k_fill_u32(unsigned* p, unsigned v, int n) {
  int i = blockIdx.x * blockDim.x + threadIdx.x;
  if (i < n) p[i] = v;
}

// ---------------- WMMA GEMM: C[M,128] (+)= A[M,128] @ W[128,128] ----------------
// block = 256 threads = 8 waves; each block does a 128x128 output panel.
// LDS holds the A panel and W in bf16, PRE-SWIZZLED into the exact per-lane
// WMMA fragment layout: [fragment][lane][16 halfs]. Fragment loads are then a
// single contiguous 32B v16bf read (2x ds_load_b128), no scalar gathers.
__global__ __launch_bounds__(256)
void k_gemm128(const float* __restrict__ A, const float* __restrict__ W,
               const float* __restrict__ bias, float* __restrict__ C,
               int M, int addC, int relu) {
  __shared__ alignas(32) __bf16 sA[DIM * DIM];  // [wave*4+kk][lane][h]
  __shared__ alignas(32) __bf16 sB[DIM * DIM];  // [nt*4+kk][lane][h]
  const int tid  = threadIdx.x;
  const int row0 = blockIdx.x * DIM;

  // Prefetch next A panel while we work on this one (global_prefetch_b8).
  if (row0 + DIM < M) __builtin_prefetch(&A[(size_t)(row0 + DIM) * DIM], 0, 1);

  for (int i = tid; i < DIM * DIM; i += 256) {
    // ---- W: i = k*128 + n  (coalesced global read, swizzled LDS store) ----
    {
      const int k = i >> 7, n = i & 127;
      const int kk = k >> 5, krem = k & 31;
      const int hi = krem >> 4, h = krem & 15;      // B layout: k = kk*32 + h + hi*16
      const int nt = n >> 4,  lh = n & 15;
      sB[(((nt * 4 + kk) * 32) + hi * 16 + lh) * 16 + h] = (__bf16)W[i];
    }
    // ---- A: i = r*128 + c  (coalesced global read, swizzled LDS store) ----
    {
      const int r = i >> 7, c = i & 127;
      const int gr = row0 + r;
      const float av = (gr < M) ? A[(size_t)gr * DIM + c] : 0.0f;
      const int kk = c >> 5, crem = c & 31;
      const int hi = crem >> 4;                     // A layout: k = kk*32 + h + hi*16
      const int h  = crem & 15;
      const int wv = r >> 4, lh = r & 15;
      sA[(((wv * 4 + kk) * 32) + hi * 16 + lh) * 16 + h] = (__bf16)av;
    }
  }
  __syncthreads();

  const int wave = tid >> 5;
  const int lane = tid & 31;
  const int lh   = lane & 15;   // col-in-tile for B/D
  const int hi   = lane >> 4;

  const v16bf* sAv = (const v16bf*)sA;
  const v16bf* sBv = (const v16bf*)sB;

  v16bf afrag[4];
#pragma unroll
  for (int kk = 0; kk < 4; ++kk) afrag[kk] = sAv[(wave * 4 + kk) * 32 + lane];

#pragma unroll
  for (int nt = 0; nt < 8; ++nt) {
    v8f acc = {};
#pragma unroll
    for (int kk = 0; kk < 4; ++kk) {
      const v16bf bfrag = sBv[(nt * 4 + kk) * 32 + lane];
      acc = __builtin_amdgcn_wmma_f32_16x16x32_bf16(
          false, afrag[kk], false, bfrag, (short)0, acc, false, false);
    }
    // D layout: lane col n=lh; VGPR r -> row r (lanes0-15) / r+8 (lanes16-31).
    const int n = nt * 16 + lh;
#pragma unroll
    for (int r = 0; r < 8; ++r) {
      const int m = row0 + wave * 16 + hi * 8 + r;
      if (m < M) {
        const size_t idx = (size_t)m * DIM + n;
        float v = acc[r];
        if (bias) v += bias[n];
        if (addC) v += C[idx];
        if (relu) v = v > 0.f ? v : 0.f;
        C[idx] = v;
      }
    }
  }
}

// ---------------- GAT edge phases ----------------
__global__ void k_rowdot2(const float* __restrict__ h, const float* __restrict__ as,
                          const float* __restrict__ ad, float* __restrict__ es,
                          float* __restrict__ ed) {
  int i = blockIdx.x * blockDim.x + threadIdx.x;
  if (i >= NNODES) return;
  const float* row = h + (size_t)i * DIM;
  float s0 = 0.f, s1 = 0.f;
  for (int f = 0; f < DIM; ++f) { float v = row[f]; s0 += v * as[f]; s1 += v * ad[f]; }
  es[i] = s0; ed[i] = s1;
}

__global__ void k_edge_logit_max(const float* __restrict__ es, const float* __restrict__ ed,
                                 const float* __restrict__ mask, const int* __restrict__ src,
                                 const int* __restrict__ dst, unsigned* __restrict__ menc) {
  int idx = blockIdx.x * blockDim.x + threadIdx.x;
  if (idx >= NEDGES + NNODES) return;
  int s, d; float ev;
  if (idx < NEDGES) { s = src[idx]; d = dst[idx]; ev = mask[s] * mask[d]; }
  else              { s = d = idx - NEDGES;       ev = mask[s]; }
  float l = d_lrelu02(es[s] + ed[d]);
  if (ev <= 0.f) l = NEGV;
  atomicMax(&menc[d], d_f2ord(l));
}

__global__ void k_edge_exp_sum(const float* __restrict__ es, const float* __restrict__ ed,
                               const float* __restrict__ mask, const int* __restrict__ src,
                               const int* __restrict__ dst, const unsigned* __restrict__ menc,
                               float* __restrict__ p, float* __restrict__ denom) {
  int idx = blockIdx.x * blockDim.x + threadIdx.x;
  if (idx >= NEDGES + NNODES) return;
  int s, d; float ev;
  if (idx < NEDGES) { s = src[idx]; d = dst[idx]; ev = mask[s] * mask[d]; }
  else              { s = d = idx - NEDGES;       ev = mask[s]; }
  float pv = 0.f;
  if (ev > 0.f) {
    const float l = d_lrelu02(es[s] + ed[d]);
    pv = ev * expf(l - d_ord2f(menc[d]));
  }
  p[idx] = pv;
  if (pv != 0.f) atomicAdd(&denom[d], pv);
}

__global__ void k_gat_aggregate(const float* __restrict__ h, const int* __restrict__ src,
                                const int* __restrict__ dst, const float* __restrict__ p,
                                const float* __restrict__ denom, float* __restrict__ out) {
  int t = blockIdx.x * blockDim.x + threadIdx.x;
  if (t >= (NEDGES + NNODES) * 32) return;
  const int e = t >> 5, f = (t & 31) << 2;
  const float pv = p[e];
  if (pv == 0.f) return;
  int s, d;
  if (e < NEDGES) { s = src[e]; d = dst[e]; } else { s = d = e - NEDGES; }
  const float dn   = denom[d];
  const float coef = pv / (dn > 0.f ? dn : 1.f);
  const float4 hv  = *(const float4*)(h + (size_t)s * DIM + f);
  float* o = out + (size_t)d * DIM + f;
  atomicAdd(o + 0, coef * hv.x);
  atomicAdd(o + 1, coef * hv.y);
  atomicAdd(o + 2, coef * hv.z);
  atomicAdd(o + 3, coef * hv.w);
}

__global__ void k_bias_mask_relu(float* __restrict__ o, const float* __restrict__ bias,
                                 const float* __restrict__ mask) {
  int idx = blockIdx.x * blockDim.x + threadIdx.x;
  if (idx >= NNODES * DIM) return;
  const int f = idx & (DIM - 1);
  const int node = idx >> 7;
  float v = o[idx] + (bias ? bias[f] : 0.f);
  v *= mask[node];
  o[idx] = v > 0.f ? v : 0.f;
}

// ---------------- GraphConv gather ----------------
__global__ void k_conv_gather(const float* __restrict__ x, const float* __restrict__ mask,
                              const int* __restrict__ src, const int* __restrict__ dst,
                              float* __restrict__ agg) {
  int t = blockIdx.x * blockDim.x + threadIdx.x;
  if (t >= NEDGES * 32) return;
  const int e = t >> 5, f = (t & 31) << 2;
  const int s = src[e], d = dst[e];
  const float ev = mask[s] * mask[d];
  if (ev <= 0.f) return;
  const float4 xv = *(const float4*)(x + (size_t)s * DIM + f);
  float* o = agg + (size_t)d * DIM + f;
  atomicAdd(o + 0, xv.x * ev);
  atomicAdd(o + 1, xv.y * ev);
  atomicAdd(o + 2, xv.z * ev);
  atomicAdd(o + 3, xv.w * ev);
}

// ---------------- TopK pooling (one block per graph) ----------------
__global__ __launch_bounds__(256)
void k_topk(float* __restrict__ x, float* __restrict__ mask,
            const float* __restrict__ w, int k) {
  __shared__ float ssc[NPG];
  __shared__ float smk[NPG];
  const int b = blockIdx.x, tid = threadIdx.x;
  float nrm = 0.f;
  for (int f = 0; f < DIM; ++f) nrm += w[f] * w[f];
  nrm = sqrtf(nrm);
  const int base = b * NPG;
  for (int i = tid; i < NPG; i += 256) {
    const float* row = x + (size_t)(base + i) * DIM;
    float sc = 0.f;
    for (int f = 0; f < DIM; ++f) sc += row[f] * w[f];
    ssc[i] = sc / nrm;
    smk[i] = mask[base + i];
  }
  __syncthreads();
  for (int i = tid; i < NPG; i += 256) {
    const float mi = smk[i];
    const float si = (mi > 0.f) ? ssc[i] : -INFINITY;
    int cnt = 0;
    for (int j = 0; j < NPG; ++j) {
      const float sj = (smk[j] > 0.f) ? ssc[j] : -INFINITY;
      cnt += (sj > si) || (sj == si && j < i);  // stable descending rank
    }
    const bool keep = (cnt < k) && (mi > 0.f);
    mask[base + i] = keep ? 1.f : 0.f;
    const float scale = keep ? tanhf(ssc[i]) : 0.f;
    float* row = x + (size_t)(base + i) * DIM;
    for (int f = 0; f < DIM; ++f) row[f] *= scale;
  }
}

// ---------------- readout: g[b, 0:128]+=max, g[b,128:256]+=mean ----------------
__global__ void k_readout_add(const float* __restrict__ x, const float* __restrict__ mask,
                              float invk, float* __restrict__ g) {
  const int b = blockIdx.x, f = threadIdx.x;  // 128 threads
  float mx = NEGV, sm = 0.f;
  for (int i = 0; i < NPG; ++i) {
    const float m = mask[b * NPG + i];
    const float v = x[(size_t)(b * NPG + i) * DIM + f];
    const float vm = (m > 0.f) ? v : NEGV;
    mx = vm > mx ? vm : mx;
    sm += v * m;
  }
  g[b * 256 + f]       += mx;
  g[b * 256 + DIM + f] += sm * invk;
}

// ---------------- final MLP ----------------
__global__ void k_lin1(const float* __restrict__ g, const float* __restrict__ W,
                       const float* __restrict__ bvec, float* __restrict__ out) {
  int idx = blockIdx.x * blockDim.x + threadIdx.x;  // NGRAPH*128
  if (idx >= NGRAPH * DIM) return;
  const int r = idx >> 7, c = idx & 127;
  float s = bvec[c];
  for (int k = 0; k < 256; ++k) s += g[r * 256 + k] * W[k * DIM + c];
  out[idx] = s > 0.f ? s : 0.f;
}
__global__ void k_lin2(const float* __restrict__ a, const float* __restrict__ W,
                       const float* __restrict__ bvec, float* __restrict__ out) {
  int idx = blockIdx.x * blockDim.x + threadIdx.x;  // NGRAPH*64
  if (idx >= NGRAPH * 64) return;
  const int r = idx >> 6, c = idx & 63;
  float s = bvec[c];
  for (int k = 0; k < DIM; ++k) s += a[r * DIM + k] * W[k * 64 + c];
  out[idx] = s > 0.f ? s : 0.01f * s;
}
__global__ void k_lin3_lsm(const float* __restrict__ a, const float* __restrict__ W,
                           const float* __restrict__ bvec, float* __restrict__ out) {
  __shared__ float lg[16];
  const int b = blockIdx.x, c = threadIdx.x;  // 32 threads
  float s = 0.f;
  if (c < NCLS) {
    s = bvec[c];
    for (int k = 0; k < 64; ++k) s += a[b * 64 + k] * W[k * NCLS + c];
    lg[c] = s;
  }
  __syncthreads();
  if (c == 0) {
    float mx = lg[0];
    for (int i = 1; i < NCLS; ++i) mx = lg[i] > mx ? lg[i] : mx;
    float se = 0.f;
    for (int i = 0; i < NCLS; ++i) se += expf(lg[i] - mx);
    lg[10] = mx; lg[11] = logf(se);
  }
  __syncthreads();
  if (c < NCLS) out[b * NCLS + c] = s - lg[10] - lg[11];
}

// ---------------- host-side orchestration ----------------
static inline int cdiv(int a, int b) { return (a + b - 1) / b; }

extern "C" void kernel_launch(void* const* d_in, const int* in_sizes, int n_in,
                              void* d_out, int out_size, void* d_ws, size_t ws_size,
                              hipStream_t stream) {
  (void)in_sizes; (void)n_in; (void)out_size; (void)ws_size;
  const float* x     = (const float*)d_in[0];
  const float* W_g[3]  = {(const float*)d_in[1], (const float*)d_in[5], (const float*)d_in[9]};
  const float* as_g[3] = {(const float*)d_in[2], (const float*)d_in[6], (const float*)d_in[10]};
  const float* ad_g[3] = {(const float*)d_in[3], (const float*)d_in[7], (const float*)d_in[11]};
  const float* b_g[3]  = {(const float*)d_in[4], (const float*)d_in[8], (const float*)d_in[12]};
  const float* Wr_c[3] = {(const float*)d_in[13], (const float*)d_in[16], (const float*)d_in[19]};
  const float* br_c[3] = {(const float*)d_in[14], (const float*)d_in[17], (const float*)d_in[20]};
  const float* Wo_c[3] = {(const float*)d_in[15], (const float*)d_in[18], (const float*)d_in[21]};
  const float* w_gat[3]  = {(const float*)d_in[22], (const float*)d_in[22], (const float*)d_in[23]};
  const float* w_conv[3] = {(const float*)d_in[24], (const float*)d_in[25], (const float*)d_in[26]};
  const float* W_l1 = (const float*)d_in[27]; const float* b_l1 = (const float*)d_in[28];
  const float* W_l2 = (const float*)d_in[29]; const float* b_l2 = (const float*)d_in[30];
  const float* W_l3 = (const float*)d_in[31]; const float* b_l3 = (const float*)d_in[32];
  const int* src = (const int*)d_in[33];
  const int* dst = src + NEDGES;
  float* out = (float*)d_out;

  // Workspace carve-up
  float* buf0 = (float*)d_ws;
  float* buf1 = buf0 + (size_t)NNODES * DIM;
  float* buf2 = buf1 + (size_t)NNODES * DIM;
  float* es   = buf2 + (size_t)NNODES * DIM;
  float* ed   = es + NNODES;
  unsigned* menc = (unsigned*)(ed + NNODES);
  float* denom = (float*)(menc + NNODES);
  float* pbuf  = denom + NNODES;                 // NEDGES + NNODES
  float* mask  = pbuf + (NEDGES + NNODES);
  float* g     = mask + NNODES;                  // NGRAPH*256
  float* l1    = g + NGRAPH * 256;
  float* l2    = l1 + NGRAPH * DIM;

  const int Kp[3]   = {800, 640, 512};
  const unsigned negEnc = h_f2ord(NEGV);
  const int gemmBlocks  = cdiv(NNODES, 128);
  const int nodeFeatBlk = cdiv(NNODES * DIM, 256);
  const int edgeBlk     = cdiv(NEDGES + NNODES, 256);
  const int aggBlk      = cdiv((NEDGES + NNODES) * 32, 256);
  const int gatherBlk   = cdiv(NEDGES * 32, 256);

  // Zero readout accumulator, set node mask = 1
  k_fill_f32<<<cdiv(NGRAPH * 256, 256), 256, 0, stream>>>(g, 0.f, NGRAPH * 256);
  k_fill_f32<<<cdiv(NNODES, 256), 256, 0, stream>>>(mask, 1.f, NNODES);

  // ---- GAT branch ----
  const float* gin = x;
  float* hb = buf1;
  float* ob[3] = {buf2, buf0, buf2};
  for (int L = 0; L < 3; ++L) {
    k_gemm128<<<gemmBlocks, 256, 0, stream>>>(gin, W_g[L], nullptr, hb, NNODES, 0, 0);
    k_rowdot2<<<cdiv(NNODES, 256), 256, 0, stream>>>(hb, as_g[L], ad_g[L], es, ed);
    k_fill_u32<<<cdiv(NNODES, 256), 256, 0, stream>>>(menc, negEnc, NNODES);
    k_fill_f32<<<cdiv(NNODES, 256), 256, 0, stream>>>(denom, 0.f, NNODES);
    k_edge_logit_max<<<edgeBlk, 256, 0, stream>>>(es, ed, mask, src, dst, menc);
    k_edge_exp_sum<<<edgeBlk, 256, 0, stream>>>(es, ed, mask, src, dst, menc, pbuf, denom);
    k_fill_f32<<<nodeFeatBlk, 256, 0, stream>>>(ob[L], 0.f, NNODES * DIM);
    k_gat_aggregate<<<aggBlk, 256, 0, stream>>>(hb, src, dst, pbuf, denom, ob[L]);
    k_bias_mask_relu<<<nodeFeatBlk, 256, 0, stream>>>(ob[L], b_g[L], mask);
    k_topk<<<NGRAPH, 256, 0, stream>>>(ob[L], mask, w_gat[L], Kp[L]);
    k_readout_add<<<NGRAPH, 128, 0, stream>>>(ob[L], mask, 1.f / (float)Kp[L], g);
    gin = ob[L];
  }

  // ---- GraphConv branch (restarts from original x, fresh mask) ----
  k_fill_f32<<<cdiv(NNODES, 256), 256, 0, stream>>>(mask, 1.f, NNODES);
  const float* cin = x;
  float* agg = buf1;
  float* cb[3] = {buf2, buf0, buf2};
  for (int L = 0; L < 3; ++L) {
    k_fill_f32<<<nodeFeatBlk, 256, 0, stream>>>(agg, 0.f, NNODES * DIM);
    k_conv_gather<<<gatherBlk, 256, 0, stream>>>(cin, mask, src, dst, agg);
    k_gemm128<<<gemmBlocks, 256, 0, stream>>>(agg, Wr_c[L], br_c[L], cb[L], NNODES, 0, 0);
    k_gemm128<<<gemmBlocks, 256, 0, stream>>>(cin, Wo_c[L], nullptr, cb[L], NNODES, 1, 0);
    k_bias_mask_relu<<<nodeFeatBlk, 256, 0, stream>>>(cb[L], nullptr, mask);
    k_topk<<<NGRAPH, 256, 0, stream>>>(cb[L], mask, w_conv[L], Kp[L]);
    k_readout_add<<<NGRAPH, 128, 0, stream>>>(cb[L], mask, 1.f / (float)Kp[L], g);
    cin = cb[L];
  }

  // ---- final MLP + log_softmax ----
  k_lin1<<<cdiv(NGRAPH * DIM, 256), 256, 0, stream>>>(g, W_l1, b_l1, l1);
  k_lin2<<<cdiv(NGRAPH * 64, 256), 256, 0, stream>>>(l1, W_l2, b_l2, l2);
  k_lin3_lsm<<<NGRAPH, 32, 0, stream>>>(l2, W_l3, b_l3, out);
}
